// BahdanauAttention_16381005267014
// MI455X (gfx1250) — matrix operand order repaired and checked by reading, not compiled, not run on hardware
//
#include <hip/hip_runtime.h>
#include <hip/hip_bf16.h>

#define B_ 64
#define S_ 2048
#define H_ 1024
#define MTILE 64                     // S-rows per workgroup (16 per wave, 4 waves)
#define LDS_STRIDE (H_ + 8)          // halves; +8 padding to spread LDS banks
#define LDS_BYTES (MTILE * LDS_STRIDE * 2)

typedef __attribute__((ext_vector_type(16))) __bf16 v16bf;
typedef __attribute__((ext_vector_type(8)))  __bf16 v8bf;
typedef __attribute__((ext_vector_type(4)))  __bf16 v4bf;
typedef __attribute__((ext_vector_type(8)))  float  v8f;
typedef __attribute__((ext_vector_type(4)))  float  v4f;

static __device__ __forceinline__ v16bf combine16(v8bf lo, v8bf hi) {
  v16bf r;
#pragma unroll
  for (int i = 0; i < 8; ++i) { r[i] = lo[i]; r[i + 8] = hi[i]; }
  return r;
}

// branch-free tanh: (e-1)/(e+1), e = exp(2*clamp(x,-15,15))
static __device__ __forceinline__ float fast_tanh(float x) {
  float xc = fminf(fmaxf(x, -15.f), 15.f);
  float e  = __expf(2.f * xc);
  return (e - 1.f) * __builtin_amdgcn_rcpf(e + 1.f);
}

// ---------------- Kernel 1: W_enc f32 -> bf16 (2 MiB, lives in L2) ----------
__global__ void cvt_f32_to_bf16_kernel(const float* __restrict__ src,
                                       __bf16* __restrict__ dst, int n4) {
  int i = blockIdx.x * blockDim.x + threadIdx.x;
  if (i < n4) {
    v4f x = ((const v4f*)src)[i];
    v4bf y = { (__bf16)x.x, (__bf16)x.y, (__bf16)x.z, (__bf16)x.w };
    ((v4bf*)dst)[i] = y;
  }
}

// ---------------- Kernel 2: dec_proj[b][o] = dec[b,:] . W_dec[o,:] ----------
__global__ void dec_proj_kernel(const float* __restrict__ dec,
                                const float* __restrict__ Wdec,
                                float* __restrict__ out) {
  int o = blockIdx.x * blockDim.x + threadIdx.x;   // 0..H-1
  int b = blockIdx.y;                              // 0..B-1
  const float* d = dec + (size_t)b * H_;
  const float* w = Wdec + (size_t)o * H_;
  float acc = 0.f;
#pragma unroll 4
  for (int h = 0; h < H_; h += 4) {
    v4f dv = *(const v4f*)(d + h);
    v4f wv = *(const v4f*)(w + h);
    acc += dv.x * wv.x + dv.y * wv.y + dv.z * wv.z + dv.w * wv.w;
  }
  out[(size_t)b * H_ + o] = acc;
}

// ---------------- Kernel 3: fused enc@W_enc^T + dec_proj -> tanh -> .v ------
// grid: (S/MTILE, B), block: 128 threads = 4 wave32.
// Each wave owns a 16-row M-tile; A staged in LDS as bf16. Outer loop walks
// N in blocks of 32 (two 16-col tiles share one A fragment -> 2 wmma per A
// load). K loop is software-pipelined: A one stage ahead (LDS), B two stages
// ahead (L2 latency), with __builtin_amdgcn_sched_barrier(0) fences so the
// machine scheduler cannot sink the prefetch loads back under the wmmas
// (observed in the previous build: every wmma stalled on loadcnt==0).
__global__ void fused_scores_kernel(const float* __restrict__ enc,
                                    const __bf16* __restrict__ Wbf,
                                    const float* __restrict__ decProj,
                                    const float* __restrict__ vvec,
                                    float* __restrict__ scores) {
  extern __shared__ __align__(16) char smem[];
  __bf16* ldsA = (__bf16*)smem;

  const int b  = blockIdx.y;
  const int m0 = blockIdx.x * MTILE;
  const int tid = threadIdx.x;

  // ---- Stage MTILE x H of enc (f32) into LDS as bf16, coalesced float4 ----
  const float* encB = enc + ((size_t)b * S_ + m0) * H_;
  for (int idx = tid; idx < MTILE * (H_ / 4); idx += 128) {
    int r  = idx >> 8;                 // / (H/4)  == /256
    int c4 = idx & 255;                // % 256
    v4f x = *(const v4f*)(encB + (size_t)r * H_ + c4 * 4);
    v4bf y = { (__bf16)x.x, (__bf16)x.y, (__bf16)x.z, (__bf16)x.w };
    *(v4bf*)&ldsA[r * LDS_STRIDE + c4 * 4] = y;
  }
  __syncthreads();

  const int wave = tid >> 5;           // 0..3
  const int lane = tid & 31;
  const int lr   = lane & 15;          // row (A) / col (B,C) within tile
  const int hi   = lane >> 4;          // lane half selects K sub-chunk
  const int kselA = hi ? 16 : 0;       // A: lanes16-31 hold K+16..K+31
  const int kselB = hi ? 16 : 0;       // B: lanes16-31 hold K+16..K+31
  const __bf16* aRowBase = &ldsA[(wave * 16 + lr) * LDS_STRIDE];
  const size_t bH = (size_t)b * H_;

  float rsum[8];
#pragma unroll
  for (int r = 0; r < 8; ++r) rsum[r] = 0.f;

  for (int nb = 0; nb < H_; nb += 32) {
    int kopq = 0;
    asm volatile("" : "+v"(kopq));     // opaque 0: blocks LICM/PRE of A loads
    const __bf16* aR = aRowBase + kopq + kselA;
    const __bf16* w0 = Wbf + (size_t)(nb      + lr) * H_ + kselB;
    const __bf16* w1 = Wbf + (size_t)(nb + 16 + lr) * H_ + kselB;
    if (nb + 32 < H_)  // warm next W block (global_prefetch_b8 path)
      __builtin_prefetch(Wbf + (size_t)(nb + 32 + lr) * H_, 0, 1);

    // epilogue operands in flight during the MMA loop
    float dp0 = decProj[bH + nb + lr];
    float vv0 = vvec[nb + lr];
    float dp1 = decProj[bH + nb + 16 + lr];
    float vv1 = vvec[nb + 16 + lr];

    v8f c0 = {}, c1 = {};
    // pipeline prologue: A(k=0); B stage0 (k=0) and stage1 (k=32)
    v8bf a0 = *(const v8bf*)(aR);
    v8bf a1 = *(const v8bf*)(aR + 8);
    v8bf p0 = *(const v8bf*)(w0);
    v8bf p1 = *(const v8bf*)(w0 + 8);
    v8bf q0 = *(const v8bf*)(w1);
    v8bf q1 = *(const v8bf*)(w1 + 8);
    v8bf p2 = *(const v8bf*)(w0 + 32);
    v8bf p3 = *(const v8bf*)(w0 + 40);
    v8bf q2 = *(const v8bf*)(w1 + 32);
    v8bf q3 = *(const v8bf*)(w1 + 40);

#pragma unroll 4
    for (int k0 = 0; k0 < H_; k0 += 32) {
      // issue: A for k+32, B for k+64. Trailing iterations overshoot the row;
      // those loads land in valid memory (ws) / return 0 (LDS OOB) and the
      // values are never consumed -- keeps the loop branch-free.
      v8bf na0 = *(const v8bf*)(aR + k0 + 32);
      v8bf na1 = *(const v8bf*)(aR + k0 + 40);
      v8bf nP0 = *(const v8bf*)(w0 + k0 + 64);
      v8bf nP1 = *(const v8bf*)(w0 + k0 + 72);
      v8bf nQ0 = *(const v8bf*)(w1 + k0 + 64);
      v8bf nQ1 = *(const v8bf*)(w1 + k0 + 72);
      __builtin_amdgcn_sched_barrier(0);   // keep prefetches above the MMAs
      v16bf af = combine16(a0, a1);
      c0 = __builtin_amdgcn_wmma_f32_16x16x32_bf16(
               false, af, false, combine16(p0, p1), (short)0, c0, false, false);
      c1 = __builtin_amdgcn_wmma_f32_16x16x32_bf16(
               false, af, false, combine16(q0, q1), (short)0, c1, false, false);
      __builtin_amdgcn_sched_barrier(0);   // keep MMAs above the rotation
      a0 = na0; a1 = na1;
      p0 = p2; p1 = p3; q0 = q2; q1 = q3;
      p2 = nP0; p3 = nP1; q2 = nQ0; q3 = nQ1;
    }

    // epilogue: lane holds columns N=nb+lr and N=nb+16+lr for rows M=r+8*hi
#pragma unroll
    for (int r = 0; r < 8; ++r)
      rsum[r] += fast_tanh(c0[r] + dp0) * vv0 + fast_tanh(c1[r] + dp1) * vv1;
  }

  // reduce the 16 column-lanes of each half-wave; lane lr==0 writes 8 rows
#pragma unroll
  for (int r = 0; r < 8; ++r) {
    float s = rsum[r];
    s += __shfl_xor(s, 8, 16);
    s += __shfl_xor(s, 4, 16);
    s += __shfl_xor(s, 2, 16);
    s += __shfl_xor(s, 1, 16);
    if (lr == 0)
      scores[(size_t)b * S_ + m0 + wave * 16 + hi * 8 + r] = s;
  }
}

// ---------------- Kernel 4: softmax over S per batch row --------------------
__global__ void softmax_kernel(const float* __restrict__ scores,
                               float* __restrict__ weights) {
  __shared__ float red[256];
  const int b = blockIdx.x, t = threadIdx.x;
  const float* sr = scores + (size_t)b * S_;
  float* wr = weights + (size_t)b * S_;

  float m = -3.402823466e+38f;
  for (int s = t; s < S_; s += 256) m = fmaxf(m, sr[s]);
  red[t] = m; __syncthreads();
  for (int o = 128; o > 0; o >>= 1) {
    if (t < o) red[t] = fmaxf(red[t], red[t + o]);
    __syncthreads();
  }
  const float mx = red[0]; __syncthreads();

  float sum = 0.f;
  for (int s = t; s < S_; s += 256) {
    float e = __expf(sr[s] - mx);
    wr[s] = e; sum += e;
  }
  red[t] = sum; __syncthreads();
  for (int o = 128; o > 0; o >>= 1) {
    if (t < o) red[t] += red[t + o];
    __syncthreads();
  }
  const float inv = 1.f / red[0];
  for (int s = t; s < S_; s += 256) wr[s] *= inv;
}

// ---------------- Kernel 5: context partials over S chunks ------------------
// grid (8, B), block 256: thread owns 4 h; chunk = 256 s values.
__global__ void context_partial_kernel(const float* __restrict__ enc,
                                       const float* __restrict__ weights,
                                       float* __restrict__ partial) {
  const int sc = blockIdx.x, b = blockIdx.y;
  const int h4 = threadIdx.x;                    // h = h4*4
  const float* e = enc + ((size_t)b * S_ + (size_t)sc * 256) * H_;
  const float* w = weights + (size_t)b * S_ + sc * 256;
  v4f acc = {0.f, 0.f, 0.f, 0.f};
#pragma unroll 4
  for (int s = 0; s < 256; ++s) {
    float ws = w[s];
    v4f x = *(const v4f*)(e + (size_t)s * H_ + h4 * 4);
    acc += x * ws;
  }
  *(v4f*)(partial + ((size_t)(b * 8 + sc)) * H_ + h4 * 4) = acc;
}

__global__ void context_reduce_kernel(const float* __restrict__ partial,
                                      float* __restrict__ ctx) {
  const int b = blockIdx.x, h4 = threadIdx.x;
  v4f acc = {0.f, 0.f, 0.f, 0.f};
#pragma unroll
  for (int sc = 0; sc < 8; ++sc)
    acc += *(const v4f*)(partial + ((size_t)(b * 8 + sc)) * H_ + h4 * 4);
  *(v4f*)(ctx + (size_t)b * H_ + h4 * 4) = acc;
}

// ---------------------------------------------------------------------------
extern "C" void kernel_launch(void* const* d_in, const int* in_sizes, int n_in,
                              void* d_out, int out_size, void* d_ws, size_t ws_size,
                              hipStream_t stream) {
  (void)in_sizes; (void)n_in; (void)out_size; (void)ws_size;
  const float* enc  = (const float*)d_in[0];   // (B,S,H)
  const float* dec  = (const float*)d_in[1];   // (B,H)
  const float* Wenc = (const float*)d_in[2];   // (H,H)
  const float* Wdec = (const float*)d_in[3];   // (H,H)
  const float* v    = (const float*)d_in[4];   // (H,)

  float* ctx     = (float*)d_out;              // (B,H)
  float* weights = (float*)d_out + B_ * H_;    // (B,S)

  char* ws = (char*)d_ws;
  __bf16* Wbf     = (__bf16*)ws;                                   // 2 MiB
  float*  decProj = (float*)(ws + 2u * 1024 * 1024);               // 256 KiB
  float*  scores  = (float*)(ws + 2u * 1024 * 1024 + 256 * 1024);  // 512 KiB
  float*  partial = (float*)(ws + 2u * 1024 * 1024 + 768 * 1024);  // 2 MiB

  hipFuncSetAttribute((const void*)fused_scores_kernel,
                      hipFuncAttributeMaxDynamicSharedMemorySize, LDS_BYTES);

  cvt_f32_to_bf16_kernel<<<(H_ * H_ / 4 + 255) / 256, 256, 0, stream>>>(
      Wenc, Wbf, H_ * H_ / 4);
  dec_proj_kernel<<<dim3(H_ / 256, B_), 256, 0, stream>>>(dec, Wdec, decProj);
  fused_scores_kernel<<<dim3(S_ / MTILE, B_), 128, LDS_BYTES, stream>>>(
      enc, Wbf, decProj, v, scores);
  softmax_kernel<<<B_, 256, 0, stream>>>(scores, weights);
  context_partial_kernel<<<dim3(8, B_), 256, 0, stream>>>(enc, weights, partial);
  context_reduce_kernel<<<B_, 256, 0, stream>>>(partial, ctx);
}
